// VP_Model_46901042872500
// MI455X (gfx1250) — compile-verified, matrix operand order checked by reading, not run on hardware
//
#include <hip/hip_runtime.h>
#include <math.h>

// ---------------------------------------------------------------------------
// VP point-transformer forward for MI455X (gfx1250, wave32, WMMA).
//
// Input order (insertion-order flatten of setup_inputs()):
//   0: point_cloud (B,N,3)
//   1..4:   sa1  W0(64x6) b0 W1(128x64) b1
//   5..26:  t1   start(W,b) key(W,b) query(W,b) value(W,b) pos1(W,b)
//                pos_bn(g,be) pos2(W,b) attn1(W,b) attn_bn(g,be)
//                attn2(W,b) end(W,b)
//   27..30: sa2  W0(128x131) b0 W1(256x128) b1
//   31..52: t2   (same layout as t1)
//   53..56: sa3  W0(512x259) b0 W1(1024x512) b1
//   57..62: fc1(W,b) fc2(W,b) fc3(W,b)
// Output: (B,3) fp32, L2-normalized tanh.
// ---------------------------------------------------------------------------

typedef __attribute__((ext_vector_type(16))) _Float16 v16h;
typedef __attribute__((ext_vector_type(8)))  _Float16 v8h;
typedef __attribute__((ext_vector_type(8)))  float    v8f;
typedef __attribute__((ext_vector_type(4)))  float    v4f;

#define BN_INV 0.99999500003750f /* 1/sqrt(1+1e-5) */
#define LDT 36    /* f32 tile row stride: 144B, 16B-aligned, bank-disjoint */
#define MIDLD 520 /* f16 mid column stride: 1040B, 16B-aligned, bank-clean */

// ---------------- WMMA fragment helpers (16x16x32 f16, wave32) -------------
// Per ISA layout, lane L holds fragment row (M or N) = L%16 and two
// contiguous K runs: [kg*8, kg*8+7] and [kg*8+16, kg*8+23], kg = L/16.
// Tiles are stored K-contiguous per row (A: [m][k], B staged as [n][k]),
// so each fragment is wide b128 LDS loads + pack-convert.
__device__ __forceinline__ v16h load_frag_f32(const float* tile, int lane) {
  const int r = lane & 15;
  const int kg = (lane >> 4) << 3;
  const float* p = tile + r * LDT + kg;
  v4f a0 = *(const v4f*)(p);
  v4f a1 = *(const v4f*)(p + 4);
  v4f a2 = *(const v4f*)(p + 16);
  v4f a3 = *(const v4f*)(p + 20);
  v16h f;
#pragma unroll
  for (int e = 0; e < 4; ++e) {
    f[e]      = (_Float16)a0[e];
    f[4 + e]  = (_Float16)a1[e];
    f[8 + e]  = (_Float16)a2[e];
    f[12 + e] = (_Float16)a3[e];
  }
  return f;
}
// Fragment from f16 N-major mid tile: two b128 loads, no conversion.
__device__ __forceinline__ v16h load_frag_f16(const _Float16* tile, int kk,
                                              int lane) {
  const int c = lane & 15;
  const int kg = (lane >> 4) << 3;
  const _Float16* p = tile + c * MIDLD + kk + kg;
  v8h lo = *(const v8h*)(p);
  v8h hi = *(const v8h*)(p + 16);
  v16h f;
#pragma unroll
  for (int e = 0; e < 8; ++e) {
    f[e] = lo[e];
    f[8 + e] = hi[e];
  }
  return f;
}

// ---------------------------------------------------------------------------
// Generic per-batch GEMM: out[b,m,n] = act(bn(W[m,:] @ X[b,:,n] + bias)) (+res)
// Block: 128 threads = 4 waves, 32x32 tile (2x2 wave tiles of 16x16).
// pool16: max-pool groups of 16 consecutive columns (out has Ncols/16 cols).
// ---------------------------------------------------------------------------
__global__ __launch_bounds__(128) void gemm_kernel(
    const float* __restrict__ W, const float* __restrict__ X,
    const float* __restrict__ bias, const float* __restrict__ bnG,
    const float* __restrict__ bnB, const float* __restrict__ residual,
    float* __restrict__ out, int M, int K, int Ncols, int act, int pool16) {
  __shared__ float sA[32 * LDT];   // [m][k]
  __shared__ float sBt[32 * LDT];  // [n][k] (transposed)
  const int b = blockIdx.z;
  const int tid = threadIdx.x;
  const int lane = tid & 31;
  const int wave = tid >> 5;
  const int waveM = wave >> 1, waveN = wave & 1;
  const int m0 = blockIdx.y * 32;
  const int n0 = blockIdx.x * 32;
  const float* Xb = X + (size_t)b * K * Ncols;
  v8f acc = {};
  for (int kk = 0; kk < K; kk += 32) {
    const bool fullK = (kk + 32 <= K);
    __syncthreads();
    // ---- stage W tile (fast: unconditional b128 loads) ----
    if (fullK && (m0 + 32 <= M) && ((K & 3) == 0)) {
#pragma unroll
      for (int i = 0; i < 2; ++i) {
        int lin = tid + i * 128;
        int r = lin >> 3, c4 = (lin & 7) << 2;
        *(v4f*)&sA[r * LDT + c4] =
            *(const v4f*)&W[(size_t)(m0 + r) * K + kk + c4];
      }
    } else {
#pragma unroll
      for (int i = 0; i < 8; ++i) {
        int lin = tid + i * 128;
        int r = lin >> 5, c = lin & 31;
        int gm = m0 + r, gk = kk + c;
        sA[r * LDT + c] = (gm < M && gk < K) ? W[(size_t)gm * K + gk] : 0.0f;
      }
    }
    // ---- stage X tile transposed (coalesced b128 reads, scalar LDS writes) --
    if (fullK && (n0 + 32 <= Ncols)) {
#pragma unroll
      for (int i = 0; i < 2; ++i) {
        int lin = tid + i * 128;
        int gk = lin >> 3, n4 = (lin & 7) << 2;
        v4f v = *(const v4f*)&Xb[(size_t)(kk + gk) * Ncols + n0 + n4];
        sBt[(n4 + 0) * LDT + gk] = v[0];
        sBt[(n4 + 1) * LDT + gk] = v[1];
        sBt[(n4 + 2) * LDT + gk] = v[2];
        sBt[(n4 + 3) * LDT + gk] = v[3];
      }
    } else {
#pragma unroll
      for (int i = 0; i < 8; ++i) {
        int lin = tid + i * 128;
        int gk0 = lin >> 5, n = lin & 31;
        int gk = kk + gk0, gn = n0 + n;
        sBt[n * LDT + gk0] =
            (gk < K && gn < Ncols) ? Xb[(size_t)gk * Ncols + gn] : 0.0f;
      }
    }
    if (fullK && kk + 32 < K)  // gfx1250 global_prefetch_b8 on next K slice
      __builtin_prefetch(&Xb[(size_t)(kk + 32) * Ncols + n0 + lane], 0, 1);
    __syncthreads();
    v16h a = load_frag_f32(sA + (waveM * 16) * LDT, lane);
    v16h bf = load_frag_f32(sBt + (waveN * 16) * LDT, lane);
    acc = __builtin_amdgcn_wmma_f32_16x16x32_f16(false, a, false, bf, (short)0,
                                                 acc, false, false);
  }
  const int colT = n0 + waveN * 16;
  const int col = colT + (lane & 15);
  const int rbase = m0 + waveM * 16 + ((lane >> 4) << 3);
#pragma unroll
  for (int r = 0; r < 8; ++r) {
    int row = rbase + r;
    float v = acc[r];
    bool rok = (row < M);
    if (rok) {
      if (bias) v += bias[row];
      if (bnG) v = v * (bnG[row] * BN_INV) + bnB[row];
      if (act == 1) v = fmaxf(v, 0.0f);
      else if (act == 2) v = tanhf(v);
    }
    if (pool16) {
#pragma unroll
      for (int mm = 8; mm >= 1; mm >>= 1) v = fmaxf(v, __shfl_xor(v, mm, 16));
      if (rok && (lane & 15) == 0 && col < Ncols)
        out[((size_t)b * M + row) * (Ncols >> 4) + (colT >> 4)] = v;
    } else if (rok && col < Ncols) {
      size_t o = ((size_t)b * M + row) * Ncols + col;
      float vv = v;
      if (residual) vv += residual[o];
      out[o] = vv;
    }
  }
}

// ---------------------------------------------------------------------------
// Fused 2-layer MLP: out = act2(W2 @ relu(bn(W1 @ X + b1)) + b2), per batch.
// Intermediate (Mid<=512 x 32 cols) lives in LDS as f16, N-major -> phase-2
// B fragments are two ds_load_b128 per wmma, no HBM round-trip.
// Block: 256 threads = 8 waves (4 row-tiles x 2 col-tiles), 32 cols per block.
// ---------------------------------------------------------------------------
__global__ __launch_bounds__(256) void mlp2_kernel(
    const float* __restrict__ W1, const float* __restrict__ b1,
    const float* __restrict__ bnG, const float* __restrict__ bnB,
    const float* __restrict__ W2, const float* __restrict__ b2,
    const float* __restrict__ X, float* __restrict__ out, int K1, int Mid,
    int M2, int Ncols, int act2, int pool16) {
  __shared__ float sA[64 * LDT];        // weights [m][k]
  __shared__ float sBt[32 * LDT];       // X tile [n][k] (transposed)
  __shared__ _Float16 sMidT[32 * MIDLD];  // mid [col][midrow], f16
  const int b = blockIdx.z;
  const int n0 = blockIdx.x * 32;
  const int tid = threadIdx.x;
  const int lane = tid & 31;
  const int wave = tid >> 5;
  const int waveM = wave >> 1;  // 0..3
  const int waveN = wave & 1;
  const float* Xb = X + (size_t)b * K1 * Ncols;

  // ---- phase 1: mid = relu(bn(W1 @ X + b1)) -> sMidT ----
  for (int mid0 = 0; mid0 < Mid; mid0 += 64) {
    v8f acc = {};
    for (int kk = 0; kk < K1; kk += 32) {
      const bool fullK = (kk + 32 <= K1);
      __syncthreads();
      if (fullK && ((K1 & 3) == 0)) {
#pragma unroll
        for (int i = 0; i < 2; ++i) {
          int lin = tid + i * 256;
          int r = lin >> 3, c4 = (lin & 7) << 2;
          *(v4f*)&sA[r * LDT + c4] =
              *(const v4f*)&W1[(size_t)(mid0 + r) * K1 + kk + c4];
        }
      } else {
#pragma unroll
        for (int i = 0; i < 8; ++i) {
          int lin = tid + i * 256;
          int r = lin >> 5, c = lin & 31;
          int gm = mid0 + r, gk = kk + c;
          sA[r * LDT + c] =
              (gm < Mid && gk < K1) ? W1[(size_t)gm * K1 + gk] : 0.0f;
        }
      }
      if (fullK && (n0 + 32 <= Ncols)) {
        int gk = tid >> 3, n4 = (tid & 7) << 2;
        v4f v = *(const v4f*)&Xb[(size_t)(kk + gk) * Ncols + n0 + n4];
        sBt[(n4 + 0) * LDT + gk] = v[0];
        sBt[(n4 + 1) * LDT + gk] = v[1];
        sBt[(n4 + 2) * LDT + gk] = v[2];
        sBt[(n4 + 3) * LDT + gk] = v[3];
      } else {
#pragma unroll
        for (int i = 0; i < 4; ++i) {
          int lin = tid + i * 256;
          int gk0 = lin >> 5, n = lin & 31;
          int gk = kk + gk0, gn = n0 + n;
          sBt[n * LDT + gk0] =
              (gk < K1 && gn < Ncols) ? Xb[(size_t)gk * Ncols + gn] : 0.0f;
        }
      }
      __syncthreads();
      v16h a = load_frag_f32(sA + (waveM * 16) * LDT, lane);
      v16h bf = load_frag_f32(sBt + (waveN * 16) * LDT, lane);
      acc = __builtin_amdgcn_wmma_f32_16x16x32_f16(false, a, false, bf,
                                                   (short)0, acc, false, false);
    }
    __syncthreads();
    const int rbase = mid0 + waveM * 16 + ((lane >> 4) << 3);
    const int c = waveN * 16 + (lane & 15);
    v8h h;
#pragma unroll
    for (int r = 0; r < 8; ++r) {
      int row = rbase + r;
      float v = 0.0f;
      if (row < Mid) {
        v = acc[r] + b1[row];
        if (bnG) v = v * (bnG[row] * BN_INV) + bnB[row];
        v = fmaxf(v, 0.0f);
      }
      h[r] = (_Float16)v;
    }
    *(v8h*)&sMidT[c * MIDLD + rbase] = h;  // one b128 store per lane
  }
  __syncthreads();

  // ---- phase 2: out = act2(W2 @ mid + b2), mid read straight from LDS ----
  for (int m20 = 0; m20 < M2; m20 += 64) {
    v8f acc = {};
    for (int kk = 0; kk < Mid; kk += 32) {
      __syncthreads();
#pragma unroll
      for (int i = 0; i < 2; ++i) {
        int lin = tid + i * 256;
        int r = lin >> 3, c4 = (lin & 7) << 2;
        *(v4f*)&sA[r * LDT + c4] =
            *(const v4f*)&W2[(size_t)(m20 + r) * Mid + kk + c4];
      }
      __syncthreads();
      v16h a = load_frag_f32(sA + (waveM * 16) * LDT, lane);
      v16h bf = load_frag_f16(sMidT + (waveN * 16) * MIDLD, kk, lane);
      acc = __builtin_amdgcn_wmma_f32_16x16x32_f16(false, a, false, bf,
                                                   (short)0, acc, false, false);
    }
    const int colT = n0 + waveN * 16;
    const int col = colT + (lane & 15);
    const int rbase = m20 + waveM * 16 + ((lane >> 4) << 3);
#pragma unroll
    for (int r = 0; r < 8; ++r) {
      int row = rbase + r;
      float v = acc[r];
      bool rok = (row < M2);
      if (rok) {
        v += b2[row];
        if (act2 == 1) v = fmaxf(v, 0.0f);
      }
      if (pool16) {
#pragma unroll
        for (int mm = 8; mm >= 1; mm >>= 1) v = fmaxf(v, __shfl_xor(v, mm, 16));
        if (rok && (lane & 15) == 0 && col < Ncols)
          out[((size_t)b * M2 + row) * (Ncols >> 4) + (colT >> 4)] = v;
      } else if (rok && col < Ncols) {
        out[((size_t)b * M2 + row) * Ncols + col] = v;
      }
    }
  }
}

// ---------------------- irregular / elementwise kernels --------------------

__global__ void transpose_pc_kernel(const float* __restrict__ pc, int N,
                                    float* __restrict__ x, int B) {
  int g = blockIdx.x * blockDim.x + threadIdx.x;
  if (g >= B * 3 * N) return;
  int n = g % N, t = g / N, c = t % 3, b = t / 3;
  x[g] = pc[((size_t)b * N + n) * 3 + c];
}

// Furthest point sampling: one block per batch, xyz in (B,3,N).
__global__ __launch_bounds__(256) void fps_kernel(const float* __restrict__ xyz,
                                                  int N, int npoint,
                                                  int* __restrict__ outIdx) {
  const int b = blockIdx.x;
  const int tid = threadIdx.x;
  const float* xb = xyz + (size_t)b * 3 * N;
  __shared__ float sv[256];
  __shared__ int si[256];
  __shared__ int sLast;
  float px[8], py[8], pz[8], dd[8];
#pragma unroll
  for (int i = 0; i < 8; ++i) {
    int n = tid + i * 256;
    if (n < N) { px[i] = xb[n]; py[i] = xb[N + n]; pz[i] = xb[2 * N + n]; }
    else { px[i] = 0.f; py[i] = 0.f; pz[i] = 0.f; }
    dd[i] = 1e10f;
  }
  if (tid == 0) { outIdx[(size_t)b * npoint] = 0; sLast = 0; }
  __syncthreads();
  for (int s = 1; s < npoint; ++s) {
    int last = sLast;
    float lx = xb[last], ly = xb[N + last], lz = xb[2 * N + last];
    float bestv = -1.0f;
    int besti = 0;
#pragma unroll
    for (int i = 0; i < 8; ++i) {
      int n = tid + i * 256;
      float dx = px[i] - lx, dy = py[i] - ly, dz = pz[i] - lz;
      float d = dx * dx + dy * dy + dz * dz;
      dd[i] = fminf(dd[i], d);
      float val = (n < N) ? dd[i] : -1.0f;
      if (val > bestv) { bestv = val; besti = n; }
    }
    sv[tid] = bestv;
    si[tid] = besti;
    __syncthreads();
    for (int st = 128; st > 0; st >>= 1) {
      if (tid < st && sv[tid + st] > sv[tid]) {
        sv[tid] = sv[tid + st];
        si[tid] = si[tid + st];
      }
      __syncthreads();
    }
    if (tid == 0) { sLast = si[0]; outIdx[(size_t)b * npoint + s] = si[0]; }
    __syncthreads();
  }
}

// 16-NN per query (thread per query), coords in (B,3,*) layout.
__global__ void knn_kernel(const float* __restrict__ q,
                           const float* __restrict__ ref, int Mq, int Nref,
                           int* __restrict__ outIdx, int B) {
  int g = blockIdx.x * blockDim.x + threadIdx.x;
  if (g >= B * Mq) return;
  int b = g / Mq, m = g % Mq;
  const float* qb = q + (size_t)b * 3 * Mq;
  const float* rb = ref + (size_t)b * 3 * Nref;
  float qx = qb[m], qy = qb[Mq + m], qz = qb[2 * Mq + m];
  float bd[16];
  int bi[16];
#pragma unroll
  for (int j = 0; j < 16; ++j) { bd[j] = 3.4e38f; bi[j] = 0; }
  for (int n = 0; n < Nref; ++n) {
    float dx = rb[n] - qx, dy = rb[Nref + n] - qy, dz = rb[2 * Nref + n] - qz;
    float d = dx * dx + dy * dy + dz * dz;
    if (d < bd[15]) {
      int j = 15;
      while (j > 0 && bd[j - 1] > d) {
        bd[j] = bd[j - 1];
        bi[j] = bi[j - 1];
        --j;
      }
      bd[j] = d;
      bi[j] = n;
    }
  }
  int* ob = outIdx + ((size_t)b * Mq + m) * 16;
  for (int j = 0; j < 16; ++j) ob[j] = bi[j];
}

__global__ void gather_coords_kernel(const float* __restrict__ src,
                                     const int* __restrict__ idx, int C,
                                     int Nsrc, int Msel,
                                     float* __restrict__ out, int B) {
  int g = blockIdx.x * blockDim.x + threadIdx.x;
  if (g >= B * C * Msel) return;
  int m = g % Msel, t = g / Msel, c = t % C, b = t / C;
  out[g] = src[((size_t)b * C + c) * Nsrc + idx[(size_t)b * Msel + m]];
}

// X = concat(grouped_xyz - new_xyz, grouped points): (B, 3+Cpts, Msel*16)
__global__ void build_group_kernel(const float* __restrict__ xyz,
                                   const float* __restrict__ nxyz,
                                   const float* __restrict__ pts, int Cpts,
                                   const int* __restrict__ idx, int N, int Msel,
                                   float* __restrict__ out, int B) {
  int cols = Msel * 16;
  int Ct = 3 + Cpts;
  int g = blockIdx.x * blockDim.x + threadIdx.x;
  if (g >= B * Ct * cols) return;
  int col = g % cols, t = g / cols, c = t % Ct, b = t / Ct;
  int j = idx[(size_t)b * cols + col];
  float v;
  if (c < 3)
    v = xyz[((size_t)b * 3 + c) * N + j] -
        nxyz[((size_t)b * 3 + c) * Msel + (col >> 4)];
  else
    v = pts[((size_t)b * Cpts + (c - 3)) * N + j];
  out[g] = v;
}

__global__ void concat_all_kernel(const float* __restrict__ xyz,
                                  const float* __restrict__ pts, int Cpts,
                                  int N, float* __restrict__ out, int B) {
  int Ct = 3 + Cpts;
  int g = blockIdx.x * blockDim.x + threadIdx.x;
  if (g >= B * Ct * N) return;
  int n = g % N, t = g / N, c = t % Ct, b = t / Ct;
  out[g] = (c < 3) ? xyz[((size_t)b * 3 + c) * N + n]
                   : pts[((size_t)b * Cpts + (c - 3)) * N + n];
}

__global__ void gather_cols_kernel(const float* __restrict__ src,
                                   const int* __restrict__ idx, int C, int Nsrc,
                                   int cols, float* __restrict__ out, int B) {
  int g = blockIdx.x * blockDim.x + threadIdx.x;
  if (g >= B * C * cols) return;
  int col = g % cols, t = g / cols, c = t % C, b = t / C;
  out[g] = src[((size_t)b * C + c) * Nsrc + idx[(size_t)b * cols + col]];
}

__global__ void pos_rel_kernel(const float* __restrict__ pos,
                               const int* __restrict__ idx, int n,
                               float* __restrict__ out, int B) {
  int cols = n * 16;
  int g = blockIdx.x * blockDim.x + threadIdx.x;
  if (g >= B * 3 * cols) return;
  int col = g % cols, t = g / cols, c = t % 3, b = t / 3;
  const float* pb = pos + ((size_t)b * 3 + c) * n;
  out[g] = pb[col >> 4] - pb[idx[(size_t)b * cols + col]];
}

__global__ void attn_in_kernel(const float* __restrict__ qry,
                               const float* __restrict__ gk,
                               const float* __restrict__ pe, int D, int n,
                               float* __restrict__ out, int B) {
  int cols = n * 16;
  int g = blockIdx.x * blockDim.x + threadIdx.x;
  if (g >= B * D * cols) return;
  int col = g % cols, t = g / cols, d = t % D, b = t / D;
  out[g] = qry[((size_t)b * D + d) * n + (col >> 4)] - gk[g] + pe[g];
}

__global__ void softmax16_kernel(float* __restrict__ a, int total) {
  int g = blockIdx.x * blockDim.x + threadIdx.x;
  if (g >= total) return;
  float* p = a + (size_t)g * 16;
  float mx = p[0];
#pragma unroll
  for (int k = 1; k < 16; ++k) mx = fmaxf(mx, p[k]);
  float s = 0.f;
#pragma unroll
  for (int k = 0; k < 16; ++k) {
    float e = __expf(p[k] - mx);
    p[k] = e;
    s += e;
  }
  float inv = 1.0f / s;
#pragma unroll
  for (int k = 0; k < 16; ++k) p[k] *= inv;
}

__global__ void agg_kernel(const float* __restrict__ attn,
                           const float* __restrict__ value,
                           const float* __restrict__ pe, int total,
                           float* __restrict__ out) {
  int g = blockIdx.x * blockDim.x + threadIdx.x;
  if (g >= total) return;
  const float* ap = attn + (size_t)g * 16;
  const float* pp = pe + (size_t)g * 16;
  float v = value[g];
  float s = 0.f;
#pragma unroll
  for (int k = 0; k < 16; ++k) s += ap[k] * (v + pp[k]);
  out[g] = s;
}

__global__ void rowmax_kernel(const float* __restrict__ in, int L,
                              float* __restrict__ out, int total) {
  int g = blockIdx.x * blockDim.x + threadIdx.x;
  if (g >= total) return;
  const float* p = in + (size_t)g * L;
  float m = p[0];
  for (int i = 1; i < L; ++i) m = fmaxf(m, p[i]);
  out[g] = m;
}

__global__ void transpose2d_kernel(const float* __restrict__ in, int R, int C,
                                   float* __restrict__ out) {
  int g = blockIdx.x * blockDim.x + threadIdx.x;
  if (g >= R * C) return;
  int c = g % C, r = g / C;
  out[(size_t)c * R + r] = in[g];
}

__global__ void norm3_kernel(const float* __restrict__ o, int Ncols,
                             float* __restrict__ out) {
  int n = blockIdx.x * blockDim.x + threadIdx.x;
  if (n >= Ncols) return;
  float a = o[n], b = o[Ncols + n], c = o[2 * Ncols + n];
  float inv = rsqrtf(a * a + b * b + c * c);
  out[n * 3 + 0] = a * inv;
  out[n * 3 + 1] = b * inv;
  out[n * 3 + 2] = c * inv;
}

// ---------------------------- host-side driver -----------------------------

static inline int cdiv(int a, int b) { return (a + b - 1) / b; }

static void run_transformer(void* const* d_in, int base, const float* xin,
                            const float* pos, int n, int Cio, float* xs,
                            float* key, float* qry, float* val, int* tknn,
                            float* posrel, float* pe, float* gkAttn, float* ai,
                            float* agg, float* outp, int B,
                            hipStream_t stream) {
  auto F = [&](int o) { return (const float*)d_in[base + o]; };
  const int th = 256;
  const int cols = n * 16;
  // start / key / query / value (1x1 convs)
  gemm_kernel<<<dim3(cdiv(n, 32), 2, B), 128, 0, stream>>>(
      F(0), xin, F(1), nullptr, nullptr, nullptr, xs, 64, Cio, n, 0, 0);
  gemm_kernel<<<dim3(cdiv(n, 32), 2, B), 128, 0, stream>>>(
      F(2), xs, F(3), nullptr, nullptr, nullptr, key, 64, 64, n, 0, 0);
  gemm_kernel<<<dim3(cdiv(n, 32), 2, B), 128, 0, stream>>>(
      F(4), xs, F(5), nullptr, nullptr, nullptr, qry, 64, 64, n, 0, 0);
  gemm_kernel<<<dim3(cdiv(n, 32), 2, B), 128, 0, stream>>>(
      F(6), xs, F(7), nullptr, nullptr, nullptr, val, 64, 64, n, 0, 0);
  // self-knn on positions
  knn_kernel<<<cdiv(B * n, th), th, 0, stream>>>(pos, pos, n, n, tknn, B);
  // grouped key, relative positions
  gather_cols_kernel<<<cdiv(B * 64 * cols, th), th, 0, stream>>>(
      key, tknn, 64, n, cols, gkAttn, B);
  pos_rel_kernel<<<cdiv(B * 3 * cols, th), th, 0, stream>>>(pos, tknn, n,
                                                            posrel, B);
  // pe = pos2( relu(bn(pos1(pos_rel))) )  -- fused 2-layer MLP
  mlp2_kernel<<<dim3(cdiv(cols, 32), 1, B), 256, 0, stream>>>(
      F(8), F(9), F(10), F(11), F(12), F(13), posrel, pe, 3, 64, 64, cols, 0,
      0);
  // attn input = query - grouped_key + pe
  attn_in_kernel<<<cdiv(B * 64 * cols, th), th, 0, stream>>>(qry, gkAttn, pe,
                                                             64, n, ai, B);
  // attn = attn2( relu(bn(attn1(attn_in))) ) -- fused, 256-wide mid in LDS
  mlp2_kernel<<<dim3(cdiv(cols, 32), 1, B), 256, 0, stream>>>(
      F(14), F(15), F(16), F(17), F(18), F(19), ai, gkAttn, 64, 256, 64, cols,
      0, 0);
  // softmax over the 16 neighbors, then aggregate attn*(value+pe)
  softmax16_kernel<<<cdiv(B * 64 * n, th), th, 0, stream>>>(gkAttn,
                                                            B * 64 * n);
  agg_kernel<<<cdiv(B * 64 * n, th), th, 0, stream>>>(gkAttn, val, pe,
                                                      B * 64 * n, agg);
  // end conv + residual
  gemm_kernel<<<dim3(cdiv(n, 32), cdiv(Cio, 32), B), 128, 0, stream>>>(
      F(20), agg, F(21), nullptr, nullptr, xin, outp, Cio, 64, n, 0, 0);
}

extern "C" void kernel_launch(void* const* d_in, const int* in_sizes, int n_in,
                              void* d_out, int out_size, void* d_ws,
                              size_t ws_size, hipStream_t stream) {
  (void)in_sizes; (void)n_in; (void)out_size; (void)ws_size;
  const int B = 32, N = 2048, NP1 = 512, NP2 = 128;
  const int th = 256;
  auto F = [&](int i) { return (const float*)d_in[i]; };

  // ---- workspace layout (bump allocator, 256B aligned) ----
  char* wsb = (char*)d_ws;
  size_t off = 0;
  auto alloc = [&](size_t elems) -> void* {
    void* p = wsb + off;
    off += ((elems * 4 + 255) & ~(size_t)255);
    return p;
  };
  float* x     = (float*)alloc((size_t)B * 3 * N);
  int*   fidx1 = (int*)alloc((size_t)B * NP1);
  float* nxyz1 = (float*)alloc((size_t)B * 3 * NP1);
  int*   kidx1 = (int*)alloc((size_t)B * NP1 * 16);
  float* l1    = (float*)alloc((size_t)B * 128 * NP1);
  float* xs1   = (float*)alloc((size_t)B * 64 * NP1);
  float* key1  = (float*)alloc((size_t)B * 64 * NP1);
  float* qry1  = (float*)alloc((size_t)B * 64 * NP1);
  float* val1  = (float*)alloc((size_t)B * 64 * NP1);
  int*   tknn1 = (int*)alloc((size_t)B * NP1 * 16);
  float* agg1  = (float*)alloc((size_t)B * 64 * NP1);
  float* l1b   = (float*)alloc((size_t)B * 128 * NP1);
  int*   fidx2 = (int*)alloc((size_t)B * NP2);
  float* nxyz2 = (float*)alloc((size_t)B * 3 * NP2);
  int*   kidx2 = (int*)alloc((size_t)B * NP2 * 16);
  float* l2    = (float*)alloc((size_t)B * 256 * NP2);
  float* xs2   = (float*)alloc((size_t)B * 64 * NP2);
  float* key2  = (float*)alloc((size_t)B * 64 * NP2);
  float* qry2  = (float*)alloc((size_t)B * 64 * NP2);
  float* val2  = (float*)alloc((size_t)B * 64 * NP2);
  int*   tknn2 = (int*)alloc((size_t)B * NP2 * 16);
  float* agg2  = (float*)alloc((size_t)B * 64 * NP2);
  float* l2b   = (float*)alloc((size_t)B * 256 * NP2);
  float* gfeat = (float*)alloc((size_t)B * 1024);
  float* gT    = (float*)alloc((size_t)1024 * B);
  float* f1o   = (float*)alloc((size_t)512 * B);
  float* f2o   = (float*)alloc((size_t)128 * B);
  float* f3o   = (float*)alloc((size_t)3 * B);
  // large arenas (reused across stages)
  float* posrel = (float*)alloc((size_t)B * 3 * NP1 * 16);
  float* peB    = (float*)alloc((size_t)B * 64 * NP1 * 16);
  float* gkB    = (float*)alloc((size_t)B * 64 * NP1 * 16);
  float* aiB    = (float*)alloc((size_t)B * 64 * NP1 * 16);
  float* X1 = aiB;  // (B,6,8192)   sa1 input, dead before t1
  float* X2 = peB;  // (B,131,2048) sa2 input, dead before t2's pe
  float* X3 = gkB;  // (B,259,128)
  float* h3 = aiB;  // (B,1024,128)

  // x = transpose(point_cloud): (B,N,3) -> (B,3,N)
  transpose_pc_kernel<<<cdiv(B * 3 * N, th), th, 0, stream>>>(F(0), N, x, B);

  // ---- SA1: fps 512, knn 16, [6 -> 64 -> 128] + maxpool ----
  fps_kernel<<<B, 256, 0, stream>>>(x, N, NP1, fidx1);
  gather_coords_kernel<<<cdiv(B * 3 * NP1, th), th, 0, stream>>>(
      x, fidx1, 3, N, NP1, nxyz1, B);
  knn_kernel<<<cdiv(B * NP1, th), th, 0, stream>>>(nxyz1, x, NP1, N, kidx1, B);
  build_group_kernel<<<cdiv(B * 6 * NP1 * 16, th), th, 0, stream>>>(
      x, nxyz1, x, 3, kidx1, N, NP1, X1, B);
  mlp2_kernel<<<dim3(cdiv(NP1 * 16, 32), 1, B), 256, 0, stream>>>(
      F(1), F(2), nullptr, nullptr, F(3), F(4), X1, l1, 6, 64, 128, NP1 * 16,
      0, 1);

  // ---- Transformer 1 (n=512, Cio=128) ----
  run_transformer(d_in, 5, l1, nxyz1, NP1, 128, xs1, key1, qry1, val1, tknn1,
                  posrel, peB, gkB, aiB, agg1, l1b, B, stream);

  // ---- SA2: fps 128, knn 16, [131 -> 128 -> 256] + maxpool ----
  fps_kernel<<<B, 256, 0, stream>>>(nxyz1, NP1, NP2, fidx2);
  gather_coords_kernel<<<cdiv(B * 3 * NP2, th), th, 0, stream>>>(
      nxyz1, fidx2, 3, NP1, NP2, nxyz2, B);
  knn_kernel<<<cdiv(B * NP2, th), th, 0, stream>>>(nxyz2, nxyz1, NP2, NP1,
                                                   kidx2, B);
  build_group_kernel<<<cdiv(B * 131 * NP2 * 16, th), th, 0, stream>>>(
      nxyz1, nxyz2, l1b, 128, kidx2, NP1, NP2, X2, B);
  mlp2_kernel<<<dim3(cdiv(NP2 * 16, 32), 1, B), 256, 0, stream>>>(
      F(27), F(28), nullptr, nullptr, F(29), F(30), X2, l2, 131, 128, 256,
      NP2 * 16, 0, 1);

  // ---- Transformer 2 (n=128, Cio=256) ----
  run_transformer(d_in, 31, l2, nxyz2, NP2, 256, xs2, key2, qry2, val2, tknn2,
                  posrel, peB, gkB, aiB, agg2, l2b, B, stream);

  // ---- SA3 (group-all): [259 -> 512 -> 1024], maxpool over all 128 ----
  concat_all_kernel<<<cdiv(B * 259 * NP2, th), th, 0, stream>>>(nxyz2, l2b, 256,
                                                                NP2, X3, B);
  mlp2_kernel<<<dim3(cdiv(NP2, 32), 1, B), 256, 0, stream>>>(
      F(53), F(54), nullptr, nullptr, F(55), F(56), X3, h3, 259, 512, 1024,
      NP2, 0, 0);
  rowmax_kernel<<<cdiv(B * 1024, th), th, 0, stream>>>(h3, NP2, gfeat,
                                                       B * 1024);

  // ---- FC head: 1024 -> 512 -> 128 -> 3, tanh, L2-normalize ----
  transpose2d_kernel<<<cdiv(B * 1024, th), th, 0, stream>>>(gfeat, B, 1024, gT);
  gemm_kernel<<<dim3(cdiv(B, 32), cdiv(512, 32), 1), 128, 0, stream>>>(
      F(57), gT, F(58), nullptr, nullptr, nullptr, f1o, 512, 1024, B, 1, 0);
  gemm_kernel<<<dim3(1, cdiv(128, 32), 1), 128, 0, stream>>>(
      F(59), f1o, F(60), nullptr, nullptr, nullptr, f2o, 128, 512, B, 1, 0);
  gemm_kernel<<<dim3(1, 1, 1), 128, 0, stream>>>(
      F(61), f2o, F(62), nullptr, nullptr, nullptr, f3o, 3, 128, B, 2, 0);
  norm3_kernel<<<1, 64, 0, stream>>>(f3o, B, (float*)d_out);
}
